// FMNISTnet_25615184954204
// MI455X (gfx1250) — matrix operand order repaired and checked by reading, not compiled, not run on hardware
//
#include <hip/hip_runtime.h>

// ---------------------------------------------------------------------------
// FMNIST NOSO spiking net forward, 16 timesteps, B=128.
// conv2 / fc1 / fc2 use v_wmma_f32_16x16x32_f16 (f16 operands, f32 accum).
// All elementwise neuron dynamics fused into producer kernels.
// ---------------------------------------------------------------------------

typedef __attribute__((ext_vector_type(16))) _Float16 v16h;
typedef __attribute__((ext_vector_type(8)))  _Float16 v8h;
typedef __attribute__((ext_vector_type(8)))  float    v8f;

#define DM_F 0.95122945f   // exp(-1/20)
#define DS_F 0.81873077f   // exp(-1/5)
#define BATCH 128

static __device__ __forceinline__ v16h cat16(v8h lo, v8h hi) {
  return __builtin_shufflevector(lo, hi, 0,1,2,3,4,5,6,7,8,9,10,11,12,13,14,15);
}

static __device__ __forceinline__ float waveReduce(float v) {
#pragma unroll
  for (int off = 16; off > 0; off >>= 1)
    v += __shfl_down(v, off, 32);
  return v;
}

// ------------------------------- init kernels ------------------------------

__global__ void k_fillf(float* p, float v, int n) {
  int i = blockIdx.x * blockDim.x + threadIdx.x;
  if (i < n) p[i] = v;
}

__global__ void k_fillh(_Float16* p, int n) {
  int i = blockIdx.x * blockDim.x + threadIdx.x;
  if (i < n) p[i] = (_Float16)0.0f;
}

__global__ void k_cvt_w2(const float* __restrict__ w, _Float16* __restrict__ o) {
  int i = blockIdx.x * blockDim.x + threadIdx.x;      // 64*800
  if (i < 64 * 800) o[i] = (_Float16)w[i];
}

__global__ void k_cvt_wfc1(const float* __restrict__ w, _Float16* __restrict__ o) {
  int i = blockIdx.x * blockDim.x + threadIdx.x;      // 608*3136 (M padded)
  if (i < 608 * 3136) {
    int m = i / 3136, k = i - m * 3136;
    o[i] = (m < 600) ? (_Float16)w[m * 3136 + k] : (_Float16)0.0f;
  }
}

__global__ void k_cvt_wfc2(const float* __restrict__ w, _Float16* __restrict__ o) {
  int i = blockIdx.x * blockDim.x + threadIdx.x;      // 16*608 (M,K padded)
  if (i < 16 * 608) {
    int m = i / 608, k = i - m * 608;
    o[i] = (m < 10 && k < 600) ? (_Float16)w[m * 600 + k] : (_Float16)0.0f;
  }
}

// --------------------------- per-step kernels ------------------------------

// Input encoding neuron + conv1 membrane accumulation (same (B,1,28,28) shape)
__global__ void k_encode(const float* __restrict__ inp, float* input_u, float* in_spike,
                         float* vm1, float* vs1, float* sum_sp) {
  int i = blockIdx.x * blockDim.x + threadIdx.x;
  float sp = 0.0f;
  if (i < BATCH * 784) {
    float x = inp[i] * 0.3f;
    float u = input_u[i] * DM_F * (1.0f - in_spike[i]) + x;
    sp = (u - 1.0f) > 0.0f ? 1.0f : 0.0f;
    input_u[i] = u;
    in_spike[i] = sp;
    vm1[i] = vm1[i] * DM_F + sp;
    vs1[i] = vs1[i] * DS_F + sp;
  }
  float s = waveReduce(sp);
  if ((threadIdx.x & 31) == 0 && s != 0.0f) atomicAdd(sum_sp, s);
}

// conv1 (1->32, 5x5, pad 2) on vm1 and vs1 + NOSO + te/lat update
__global__ void k_conv1(const float* __restrict__ w1, const float* __restrict__ vm1,
                        const float* __restrict__ vs1, float* sav1, float* te1,
                        float* lat1, float* spike1, float* sum_sp) {
  int i = blockIdx.x * blockDim.x + threadIdx.x;      // B*32*784
  float sp = 0.0f;
  if (i < BATCH * 32 * 784) {
    int x = i % 28, y = (i / 28) % 28, oc = (i / 784) % 32, b = i / (784 * 32);
    const float* wv = w1 + oc * 25;
    const float* pm = vm1 + b * 784;
    const float* ps = vs1 + b * 784;
    float um = 0.0f, us = 0.0f;
#pragma unroll
    for (int ky = 0; ky < 5; ky++) {
      int iy = y + ky - 2;
      if ((unsigned)iy < 28u) {
#pragma unroll
        for (int kx = 0; kx < 5; kx++) {
          int ix = x + kx - 2;
          if ((unsigned)ix < 28u) {
            float w = wv[ky * 5 + kx];
            um = fmaf(w, pm[iy * 28 + ix], um);
            us = fmaf(w, ps[iy * 28 + ix], us);
          }
        }
      }
    }
    float sav = sav1[i];
    float u = sav * (um - us);
    sp = (u - 1.0f) > 0.0f ? 1.0f : 0.0f;
    sav1[i] = sav * (1.0f - sp);
    float te = te1[i];
    te += ((te != 0.0f) || (um != 0.0f)) ? 1.0f : 0.0f;
    te1[i] = te;
    lat1[i] += sp * (te - 15.0f);
    spike1[i] = sp;
  }
  float s = waveReduce(sp);
  if ((threadIdx.x & 31) == 0 && s != 0.0f) atomicAdd(sum_sp, s);
}

// latency min-pool 2x2 on (B,32,28,28) -> update conv2 membranes (B,32,14,14)
// and emit zero-padded f16 images [2][B][32][18][18] for the WMMA conv2.
__global__ void k_pool1(const float* __restrict__ spike1, const float* __restrict__ lat1,
                        float* vm2, float* vs2, _Float16* __restrict__ in2pad) {
  int i = blockIdx.x * blockDim.x + threadIdx.x;      // B*32*196
  if (i >= BATCH * 32 * 196) return;
  int x = i % 14, y = (i / 14) % 14, c = (i / 196) % 32, b = i / (196 * 32);
  int base = (b * 32 + c) * 784 + (2 * y) * 28 + 2 * x;
  float s00 = spike1[base],      s01 = spike1[base + 1];
  float s10 = spike1[base + 28], s11 = spike1[base + 29];
  float l00 = lat1[base],        l01 = lat1[base + 1];
  float l10 = lat1[base + 28],   l11 = lat1[base + 29];
  float lmin = fminf(fminf(l00, l01), fminf(l10, l11));
  float p = fmaxf(fmaxf(l00 == lmin ? s00 : 0.0f, l01 == lmin ? s01 : 0.0f),
                  fmaxf(l10 == lmin ? s10 : 0.0f, l11 == lmin ? s11 : 0.0f));
  int o = (b * 32 + c) * 196 + y * 14 + x;
  float vm = vm2[o] * DM_F + p;  vm2[o] = vm;
  float vs = vs2[o] * DS_F + p;  vs2[o] = vs;
  int pidx = (b * 32 + c) * 324 + (y + 2) * 18 + (x + 2);
  in2pad[pidx] = (_Float16)vm;
  in2pad[BATCH * 32 * 324 + pidx] = (_Float16)vs;
}

// conv2 as implicit GEMM: M=64(oc) x N=25088(b*14*14) x K=800(ic*25), f16 WMMA.
// One wave per 16-position N-tile; 4 M-tiles x {vm,vs} = 8 accumulators.
// Fused NOSO epilogue (sav/te/lat/spike).
__global__ void __launch_bounds__(32)
k_conv2_wmma(const _Float16* __restrict__ w2, const _Float16* __restrict__ in2pad,
             float* sav2, float* te2, float* lat2, float* spike2, float* sum_sp) {
  int lane = threadIdx.x;
  int n = blockIdx.x * 16 + (lane & 15);              // 0..25087
  int b = n / 196, pos = n - b * 196;
  int y = pos / 14, x = pos - y * 14;
  const _Float16* pm = in2pad + (b * 32) * 324 + y * 18 + x;
  const _Float16* ps = pm + BATCH * 32 * 324;
  const int koff  = (lane < 16) ? 0 : 16;             // unused; B fragment also uses abase
  const int abase = (lane < 16) ? 0 : 8;              // A-fragment K sub-block
  const int arow  = lane & 15;

  v8f accm[4] = {v8f{}, v8f{}, v8f{}, v8f{}};
  v8f accs[4] = {v8f{}, v8f{}, v8f{}, v8f{}};

  for (int c = 0; c < 25; ++c) {
    int k0 = c * 32;
    v16h bm{}, bs{};
#pragma unroll
    for (int j = 0; j < 16; ++j) {
      int k  = k0 + abase + j + (j >= 8 ? 8 : 0);
      int ic = k / 25;
      int t  = k - ic * 25;
      int ky = t / 5, kx = t - ky * 5;
      int idx = ic * 324 + ky * 18 + kx;              // (y+ky)*18+(x+kx) incl. pad
      bm[j] = pm[idx];
      bs[j] = ps[idx];
    }
#pragma unroll
    for (int mt = 0; mt < 4; ++mt) {
      const _Float16* wp = w2 + (mt * 16 + arow) * 800 + k0 + abase;
      v16h a = cat16(*(const v8h*)wp, *(const v8h*)(wp + 16));
      accm[mt] = __builtin_amdgcn_wmma_f32_16x16x32_f16(false, a, false, bm,
                                                        (short)0, accm[mt], false, false);
      accs[mt] = __builtin_amdgcn_wmma_f32_16x16x32_f16(false, a, false, bs,
                                                        (short)0, accs[mt], false, false);
    }
  }

  float lsum = 0.0f;
  const int mAdd = (lane < 16) ? 0 : 8;
#pragma unroll
  for (int mt = 0; mt < 4; ++mt) {
#pragma unroll
    for (int r = 0; r < 8; ++r) {
      int m = mt * 16 + r + mAdd;
      int o = (b * 64 + m) * 196 + pos;
      float um = accm[mt][r], us = accs[mt][r];
      float sav = sav2[o];
      float u = sav * (um - us);
      float sp = (u - 1.0f) > 0.0f ? 1.0f : 0.0f;
      sav2[o] = sav * (1.0f - sp);
      float te = te2[o];
      te += ((te != 0.0f) || (um != 0.0f)) ? 1.0f : 0.0f;
      te2[o] = te;
      lat2[o] += sp * (te - 15.0f);
      spike2[o] = sp;
      lsum += sp;
    }
  }
  lsum = waveReduce(lsum);
  if (lane == 0 && lsum != 0.0f) atomicAdd(sum_sp, lsum);
}

// latency min-pool on (B,64,14,14) -> fc1 membranes (B,3136), emit f16 (vm-vs)
__global__ void k_pool2(const float* __restrict__ spike2, const float* __restrict__ lat2,
                        float* vmf1, float* vsf1, _Float16* __restrict__ dfc1) {
  int i = blockIdx.x * blockDim.x + threadIdx.x;      // B*64*49
  if (i >= BATCH * 64 * 49) return;
  int x = i % 7, y = (i / 7) % 7, c = (i / 49) % 64, b = i / (49 * 64);
  int base = (b * 64 + c) * 196 + (2 * y) * 14 + 2 * x;
  float s00 = spike2[base],      s01 = spike2[base + 1];
  float s10 = spike2[base + 14], s11 = spike2[base + 15];
  float l00 = lat2[base],        l01 = lat2[base + 1];
  float l10 = lat2[base + 14],   l11 = lat2[base + 15];
  float lmin = fminf(fminf(l00, l01), fminf(l10, l11));
  float p = fmaxf(fmaxf(l00 == lmin ? s00 : 0.0f, l01 == lmin ? s01 : 0.0f),
                  fmaxf(l10 == lmin ? s10 : 0.0f, l11 == lmin ? s11 : 0.0f));
  int o = b * 3136 + c * 49 + y * 7 + x;
  float vm = vmf1[o] * DM_F + p;  vmf1[o] = vm;
  float vs = vsf1[o] * DS_F + p;  vsf1[o] = vs;
  dfc1[o] = (_Float16)(vm - vs);
}

// fc1 GEMM: u_pre = (vm-vs) @ w_fc1^T ; M=608(pad of 600), N=128, K=3136.
// Fused NOSO epilogue -> hf1 spikes.
__global__ void __launch_bounds__(32)
k_fc1_wmma(const _Float16* __restrict__ w, const _Float16* __restrict__ dfc1,
           float* savf1, float* spikef1, float* sum_sp) {
  int lane = threadIdx.x;
  int mt = blockIdx.x;                                // 0..37
  int nt = blockIdx.y;                                // 0..7
  const int koff  = (lane < 16) ? 0 : 16;
  const int abase = (lane < 16) ? 0 : 8;
  const _Float16* wr = w    + (mt * 16 + (lane & 15)) * 3136;
  const _Float16* xr = dfc1 + (nt * 16 + (lane & 15)) * 3136;
  v8f acc = v8f{};
  for (int c = 0; c < 98; ++c) {
    int k0 = c * 32;
    v16h a = cat16(*(const v8h*)(wr + k0 + abase), *(const v8h*)(wr + k0 + 16 + abase));
    v16h bf = cat16(*(const v8h*)(xr + k0 + abase), *(const v8h*)(xr + k0 + 16 + abase));
    acc = __builtin_amdgcn_wmma_f32_16x16x32_f16(false, a, false, bf,
                                                 (short)0, acc, false, false);
  }
  float lsum = 0.0f;
  const int mAdd = (lane < 16) ? 0 : 8;
  const int bcol = nt * 16 + (lane & 15);
#pragma unroll
  for (int r = 0; r < 8; ++r) {
    int m = mt * 16 + r + mAdd;
    if (m < 600) {
      int o = bcol * 600 + m;
      float sav = savf1[o];
      float u = sav * acc[r];
      float sp = (u - 1.0f) > 0.0f ? 1.0f : 0.0f;
      savf1[o] = sav * (1.0f - sp);
      spikef1[o] = sp;
      lsum += sp;
    }
  }
  lsum = waveReduce(lsum);
  if (lane == 0 && lsum != 0.0f) atomicAdd(sum_sp, lsum);
}

// fc2 membrane update from hf1 spikes; emit f16 (vm-vs) with K padded to 608
__global__ void k_fc2pre(const float* __restrict__ spikef1, float* vmf2, float* vsf2,
                         _Float16* __restrict__ dfc2) {
  int i = blockIdx.x * blockDim.x + threadIdx.x;      // B*600
  if (i >= BATCH * 600) return;
  int b = i / 600, m = i - b * 600;
  float sp = spikef1[i];
  float vm = vmf2[i] * DM_F + sp;  vmf2[i] = vm;
  float vs = vsf2[i] * DS_F + sp;  vsf2[i] = vs;
  dfc2[b * 608 + m] = (_Float16)(vm - vs);
}

// fc2 GEMM: M=16(pad of 10), N=128, K=608. Fused output-neuron epilogue:
// sav gate, t_elapse counter, out_t / out_u accumulation.
__global__ void __launch_bounds__(32)
k_fc2_wmma(const _Float16* __restrict__ w, const _Float16* __restrict__ dfc2,
           float* savf2, float* t_el, float* out_t, float* out_u, float* sum_sp) {
  int lane = threadIdx.x;
  int nt = blockIdx.x;                                // 0..7
  const int koff  = (lane < 16) ? 0 : 16;
  const int abase = (lane < 16) ? 0 : 8;
  const _Float16* wr = w    + (lane & 15) * 608;
  const _Float16* xr = dfc2 + (nt * 16 + (lane & 15)) * 608;
  v8f acc = v8f{};
  for (int c = 0; c < 19; ++c) {
    int k0 = c * 32;
    v16h a = cat16(*(const v8h*)(wr + k0 + abase), *(const v8h*)(wr + k0 + 16 + abase));
    v16h bf = cat16(*(const v8h*)(xr + k0 + abase), *(const v8h*)(xr + k0 + 16 + abase));
    acc = __builtin_amdgcn_wmma_f32_16x16x32_f16(false, a, false, bf,
                                                 (short)0, acc, false, false);
  }
  float lsum = 0.0f;
  const int mAdd = (lane < 16) ? 0 : 8;
  const int bcol = nt * 16 + (lane & 15);
#pragma unroll
  for (int r = 0; r < 8; ++r) {
    int m = r + mAdd;
    if (m < 10) {
      int o = bcol * 10 + m;
      float sav = savf2[o];
      float u = sav * acc[r];
      float sp = (u - 1.0f) > 0.0f ? 1.0f : 0.0f;
      savf2[o] = sav * (1.0f - sp);
      float te = t_el[o];
      te += ((te != 0.0f) || (u != 0.0f)) ? 1.0f : 0.0f;
      t_el[o] = te;
      out_t[o] += sp * (te - 16.0f);
      out_u[o] += sp * u;
      lsum += sp;
    }
  }
  lsum = waveReduce(lsum);
  if (lane == 0 && lsum != 0.0f) atomicAdd(sum_sp, lsum);
}

// ------------------------------- host side ---------------------------------

static inline int G(int n, int b = 256) { return (n + b - 1) / b; }

extern "C" void kernel_launch(void* const* d_in, const int* in_sizes, int n_in,
                              void* d_out, int out_size, void* d_ws, size_t ws_size,
                              hipStream_t stream) {
  const float* inp   = (const float*)d_in[0];   // (128,1,28,28)
  const float* w1    = (const float*)d_in[1];   // (32,1,5,5)
  const float* w2    = (const float*)d_in[2];   // (64,32,5,5)
  const float* wfc1  = (const float*)d_in[3];   // (600,3136)
  const float* wfc2  = (const float*)d_in[4];   // (10,600)
  float* out = (float*)d_out;                   // out_t(1280) | out_u(1280) | sum_sp(5)

  // ---- workspace layout (floats first, then f16; ~100 MB total) ----
  float* f = (float*)d_ws;
  size_t off = 0;
  auto A = [&](size_t n) { float* p = f + off; off += n; return p; };
  float* input_u = A(100352);  float* in_spike = A(100352);
  float* vm1 = A(100352);      float* vs1 = A(100352);
  float* sav1 = A(3211264);    float* te1 = A(3211264);
  float* lat1 = A(3211264);    float* spike1 = A(3211264);
  float* vm2 = A(802816);      float* vs2 = A(802816);
  float* sav2 = A(1605632);    float* te2 = A(1605632);
  float* lat2 = A(1605632);    float* spike2 = A(1605632);
  float* vmf1 = A(401408);     float* vsf1 = A(401408);
  float* savf1 = A(76800);     float* spikef1 = A(76800);
  float* vmf2 = A(76800);      float* vsf2 = A(76800);
  float* savf2 = A(1280);      float* t_el = A(1280);
  _Float16* h = (_Float16*)(f + off);
  size_t ho = 0;
  auto H = [&](size_t n) { _Float16* p = h + ho; ho += n; return p; };
  _Float16* w2h   = H(64 * 800);
  _Float16* wfc1h = H(608 * 3136);
  _Float16* wfc2h = H(16 * 608);
  _Float16* dfc1  = H(BATCH * 3136);
  _Float16* dfc2  = H(BATCH * 608);
  _Float16* in2pad = H(2 * BATCH * 32 * 324);   // [2][B][32][18][18], zero halo

  // ---- deterministic re-init every call (d_ws/d_out may be poisoned) ----
  k_fillf<<<G(401408), 256, 0, stream>>>(input_u, 0.0f, 401408);       // input_u..vs1
  k_fillf<<<G(3211264), 256, 0, stream>>>(sav1, 1.0f, 3211264);
  k_fillf<<<G(3211264), 256, 0, stream>>>(te1, 0.0f, 3211264);
  k_fillf<<<G(3211264), 256, 0, stream>>>(lat1, 16.0f, 3211264);
  k_fillf<<<G(1605632), 256, 0, stream>>>(vm2, 0.0f, 1605632);         // vm2+vs2
  k_fillf<<<G(1605632), 256, 0, stream>>>(sav2, 1.0f, 1605632);
  k_fillf<<<G(1605632), 256, 0, stream>>>(te2, 0.0f, 1605632);
  k_fillf<<<G(1605632), 256, 0, stream>>>(lat2, 16.0f, 1605632);
  k_fillf<<<G(802816), 256, 0, stream>>>(vmf1, 0.0f, 802816);          // vmf1+vsf1
  k_fillf<<<G(76800), 256, 0, stream>>>(savf1, 1.0f, 76800);
  k_fillf<<<G(153600), 256, 0, stream>>>(vmf2, 0.0f, 153600);          // vmf2+vsf2
  k_fillf<<<G(1280), 256, 0, stream>>>(savf2, 1.0f, 1280);
  k_fillf<<<G(1280), 256, 0, stream>>>(t_el, 0.0f, 1280);
  k_fillf<<<G(2565), 256, 0, stream>>>(out, 0.0f, 2565);
  k_fillh<<<G(2 * BATCH * 32 * 324), 256, 0, stream>>>(in2pad, 2 * BATCH * 32 * 324);
  k_fillh<<<G(BATCH * 608), 256, 0, stream>>>(dfc2, BATCH * 608);
  k_cvt_w2<<<G(64 * 800), 256, 0, stream>>>(w2, w2h);
  k_cvt_wfc1<<<G(608 * 3136), 256, 0, stream>>>(wfc1, wfc1h);
  k_cvt_wfc2<<<G(16 * 608), 256, 0, stream>>>(wfc2, wfc2h);

  float* s_in  = out + 2560;
  float* s_c1  = out + 2561;
  float* s_c2  = out + 2562;
  float* s_f1  = out + 2563;
  float* s_f2  = out + 2564;
  float* out_t = out;
  float* out_u = out + 1280;

  for (int step = 0; step < 16; ++step) {
    k_encode<<<G(BATCH * 784), 256, 0, stream>>>(inp, input_u, in_spike, vm1, vs1, s_in);
    k_conv1<<<G(BATCH * 32 * 784), 256, 0, stream>>>(w1, vm1, vs1, sav1, te1, lat1,
                                                     spike1, s_c1);
    k_pool1<<<G(BATCH * 32 * 196), 256, 0, stream>>>(spike1, lat1, vm2, vs2, in2pad);
    k_conv2_wmma<<<dim3(1568), dim3(32), 0, stream>>>(w2h, in2pad, sav2, te2, lat2,
                                                      spike2, s_c2);
    k_pool2<<<G(BATCH * 64 * 49), 256, 0, stream>>>(spike2, lat2, vmf1, vsf1, dfc1);
    k_fc1_wmma<<<dim3(38, 8), dim3(32), 0, stream>>>(wfc1h, dfc1, savf1, spikef1, s_f1);
    k_fc2pre<<<G(BATCH * 600), 256, 0, stream>>>(spikef1, vmf2, vsf2, dfc2);
    k_fc2_wmma<<<dim3(8), dim3(32), 0, stream>>>(wfc2h, dfc2, savf2, t_el,
                                                 out_t, out_u, s_f2);
  }
}
